// TransformerLayer_85469849190766
// MI455X (gfx1250) — compile-verified, matrix operand order checked by reading, not run on hardware
//
#include <hip/hip_runtime.h>

// ---------------------------------------------------------------------------
// Transformer encoder layer for MI455X (gfx1250, wave32, WMMA bf16,
// async global->LDS staging, double-buffered GEMM).
// B=4 S=2048 E=1024 H=16 DK=64 DFF=4096
// ---------------------------------------------------------------------------

typedef __attribute__((ext_vector_type(16))) __bf16 v16bf;
typedef __attribute__((ext_vector_type(8)))  __bf16 v8bf;
typedef __attribute__((ext_vector_type(8)))  float  v8f;

constexpr int Bc   = 4;
constexpr int Sc   = 2048;
constexpr int Ec   = 1024;
constexpr int Hc   = 16;
constexpr int DKc  = 64;
constexpr int DFFc = 4096;
constexpr int Mtok = Bc * Sc; // 8192

__device__ __forceinline__ v8f v8f_zero() {
    v8f z;
#pragma unroll
    for (int i = 0; i < 8; ++i) z[i] = 0.0f;
    return z;
}

// ---------------------------------------------------------------------------
// CDNA5 async global->LDS copy, 16 bytes per lane, tracked by ASYNCcnt.
// Generic LDS addresses carry the LDS byte offset in their low 32 bits
// (aperture check truncates, ISA 10.2), so (u32)ptr is the VDST operand.
// ---------------------------------------------------------------------------
__device__ __forceinline__ void async_copy_b128(void* lds_dst, const void* gsrc) {
    const unsigned lds = (unsigned)(uintptr_t)lds_dst;
    const unsigned long long ga = (unsigned long long)(uintptr_t)gsrc;
    asm volatile("global_load_async_to_lds_b128 %0, %1, off"
                 :: "v"(lds), "v"(ga)
                 : "memory");
}

__device__ __forceinline__ void wait_async0() {
#if __has_builtin(__builtin_amdgcn_s_wait_asynccnt)
    __builtin_amdgcn_s_wait_asynccnt(0);
#else
    asm volatile("s_wait_asynccnt 0x0" ::: "memory");
#endif
}

// Load a 16x32 bf16 A-fragment (or B-column fragment) for one lane.
// Per ISA 7.12.2: lane reads K=[c0,c0+8) and K=[c0+16,c0+24),
// c0 = (lane>=16) ? 8 : 0.  Two b128 loads.
__device__ __forceinline__ v16bf load_frag(const __bf16* rowbase, int lane) {
    const int c0 = (lane & 16) ? 8 : 0;
    const v8bf lo = *(const v8bf*)(rowbase + c0);
    const v8bf hi = *(const v8bf*)(rowbase + c0 + 16);
    v16bf r;
#pragma unroll
    for (int i = 0; i < 8; ++i) { r[i] = lo[i]; r[i + 8] = hi[i]; }
    return r;
}

__device__ __forceinline__ v8f wmma_bf16(v16bf a, v16bf b, v8f c) {
    return __builtin_amdgcn_wmma_f32_16x16x32_bf16(false, a, false, b,
                                                   (short)0, c, false, false);
}

// ---------------------------------------------------------------------------
// fp32 [K][N] -> bf16 transposed [N][K]
// ---------------------------------------------------------------------------
__global__ __launch_bounds__(256)
void w_transpose_kernel(const float* __restrict__ W, __bf16* __restrict__ Wt,
                        int K, int N) {
    size_t idx = (size_t)blockIdx.x * 256 + threadIdx.x;
    int k = (int)(idx / N);
    int n = (int)(idx % N);
    Wt[(size_t)n * K + k] = (__bf16)W[idx];
}

// ---------------------------------------------------------------------------
// LayerNorm (ddof=1) -> bf16, one row (E=1024) per workgroup of 256 threads
// ---------------------------------------------------------------------------
__global__ __launch_bounds__(256)
void layernorm_kernel(const float* __restrict__ x, const float* __restrict__ alpha,
                      const float* __restrict__ beta, __bf16* __restrict__ out) {
    const int row = blockIdx.x;
    const int t   = threadIdx.x;
    const float* xr = x + (size_t)row * Ec;
    float4 v = *(const float4*)(xr + t * 4);
    float s = v.x + v.y + v.z + v.w;
    float q = v.x * v.x + v.y * v.y + v.z * v.z + v.w * v.w;
#pragma unroll
    for (int off = 16; off > 0; off >>= 1) {
        s += __shfl_xor(s, off, 32);
        q += __shfl_xor(q, off, 32);
    }
    __shared__ float rs[8], rq[8];
    if ((t & 31) == 0) { rs[t >> 5] = s; rq[t >> 5] = q; }
    __syncthreads();
    s = 0.0f; q = 0.0f;
#pragma unroll
    for (int i = 0; i < 8; ++i) { s += rs[i]; q += rq[i]; }
    const float mean = s * (1.0f / Ec);
    const float var  = (q - s * mean) * (1.0f / (Ec - 1));
    const float inv  = 1.0f / (sqrtf(var) + 1e-6f);
    __bf16* orow = out + (size_t)row * Ec;
    const float xv[4] = {v.x, v.y, v.z, v.w};
#pragma unroll
    for (int i = 0; i < 4; ++i) {
        const int c = t * 4 + i;
        orow[c] = (__bf16)(alpha[c] * (xv[i] - mean) * inv + beta[c]);
    }
}

// ---------------------------------------------------------------------------
// Tiled bf16 WMMA GEMM: C[M,N] = A[M,K] @ B[K,N] + bias, B given transposed
// as Bt[N][K]. Block tile 128x128, BK=32, 256 threads = 8 waves (2m x 4n).
// Double-buffered LDS staging via global_load_async_to_lds_b128: issue tile
// i+1 before computing tile i; one s_wait_asynccnt + one barrier per k-step.
// MODE 0: bf16 out;  MODE 1: bf16 out + ReLU;  MODE 2: f32 out + residual.
// ---------------------------------------------------------------------------
template <int MODE>
__global__ __launch_bounds__(256)
void gemm_bf16_kernel(const __bf16* __restrict__ A, const __bf16* __restrict__ Bt,
                      const float* __restrict__ bias,
                      const float* __restrict__ residual,
                      float* __restrict__ outF, __bf16* __restrict__ outB,
                      int M, int N, int K) {
    __shared__ __align__(16) __bf16 sA[2][128 * 32];
    __shared__ __align__(16) __bf16 sB[2][128 * 32];

    const int t    = threadIdx.x;
    const int lane = t & 31;
    const int w    = t >> 5;
    const int wm   = (w & 1) * 64;
    const int wn   = (w >> 1) * 32;
    const int m0   = blockIdx.y * 128;
    const int n0   = blockIdx.x * 128;

    v8f acc[4][2];
#pragma unroll
    for (int i = 0; i < 4; ++i)
#pragma unroll
        for (int j = 0; j < 2; ++j) acc[i][j] = v8f_zero();

    const int lrow = t >> 1;          // 0..127
    const int lk   = (t & 1) * 16;    // 0 or 16
    const __bf16* gA = A  + ((size_t)(m0 + lrow) * K + lk);
    const __bf16* gB = Bt + ((size_t)(n0 + lrow) * K + lk);
    const int ldsOff = lrow * 32 + lk;

    const int nk = K >> 5;

    // prologue: stage tile 0 into buffer 0
    async_copy_b128(&sA[0][ldsOff],     gA);
    async_copy_b128(&sA[0][ldsOff + 8], gA + 8);
    async_copy_b128(&sB[0][ldsOff],     gB);
    async_copy_b128(&sB[0][ldsOff + 8], gB + 8);

    for (int i = 0; i < nk; ++i) {
        wait_async0();       // tile i landed (this wave's asyncs)
        __syncthreads();     // all waves: tile i visible, buf^1 free
        const int buf = i & 1;
        if (i + 1 < nk) {
            const int kn = (i + 1) << 5;
            async_copy_b128(&sA[buf ^ 1][ldsOff],     gA + kn);
            async_copy_b128(&sA[buf ^ 1][ldsOff + 8], gA + kn + 8);
            async_copy_b128(&sB[buf ^ 1][ldsOff],     gB + kn);
            async_copy_b128(&sB[buf ^ 1][ldsOff + 8], gB + kn + 8);
            __builtin_prefetch(gA + kn + 32, 0, 0);
            __builtin_prefetch(gB + kn + 32, 0, 0);
        }

        v16bf af[4], bfr[2];
#pragma unroll
        for (int mf = 0; mf < 4; ++mf)
            af[mf] = load_frag(&sA[buf][(wm + mf * 16 + (lane & 15)) * 32], lane);
#pragma unroll
        for (int nf = 0; nf < 2; ++nf)
            bfr[nf] = load_frag(&sB[buf][(wn + nf * 16 + (lane & 15)) * 32], lane);
#pragma unroll
        for (int mf = 0; mf < 4; ++mf)
#pragma unroll
            for (int nf = 0; nf < 2; ++nf)
                acc[mf][nf] = wmma_bf16(af[mf], bfr[nf], acc[mf][nf]);
    }

    const int g  = lane >> 4;
    const int cl = lane & 15;
#pragma unroll
    for (int mf = 0; mf < 4; ++mf)
#pragma unroll
        for (int nf = 0; nf < 2; ++nf)
#pragma unroll
            for (int r = 0; r < 8; ++r) {
                const int row = m0 + wm + mf * 16 + r + 8 * g;
                const int col = n0 + wn + nf * 16 + cl;
                float v = acc[mf][nf][r] + bias[col];
                if (MODE == 0) {
                    outB[(size_t)row * N + col] = (__bf16)v;
                } else if (MODE == 1) {
                    outB[(size_t)row * N + col] = (__bf16)fmaxf(v, 0.0f);
                } else {
                    outF[(size_t)row * N + col] =
                        residual[(size_t)row * N + col] + v;
                }
            }
}

// ---------------------------------------------------------------------------
// Flash attention (bf16 WMMA, f32 accum, online softmax).
// grid.x = B*H, grid.y = S/128. 256 threads = 8 waves; wave w owns 16 query
// rows. Key blocks of 64. K tile staged with async-to-LDS (row-major ==
// K^T columns); V tile transposed manually into LDS; P re-laid-out through
// wave-private LDS (C-layout -> A-fragment layout).
// ---------------------------------------------------------------------------
__global__ __launch_bounds__(256)
void flash_attn_kernel(const __bf16* __restrict__ Q, const __bf16* __restrict__ Kb,
                       const __bf16* __restrict__ Vb, const int* __restrict__ mask,
                       __bf16* __restrict__ ctx) {
    __shared__ __align__(16) __bf16 sK[64 * 64];        // [key][dk]
    __shared__ __align__(16) __bf16 sV[64 * 64];        // [dk][key] (transposed)
    __shared__ __align__(16) __bf16 sP[8 * 16 * 64];    // per-wave P tile

    const int t    = threadIdx.x;
    const int lane = t & 31;
    const int w    = t >> 5;
    const int b    = blockIdx.x / Hc;
    const int h    = blockIdx.x % Hc;
    const int q0   = blockIdx.y * 128 + w * 16;
    const int g    = lane >> 4;
    const int cl   = lane & 15;

    // Q A-fragments straight from global (row = q0+cl, 2 k-tiles of 32)
    v16bf qa[2];
    {
        const __bf16* qrow = Q + ((size_t)(b * Sc + q0 + cl) * Ec + h * DKc);
#pragma unroll
        for (int kt = 0; kt < 2; ++kt) qa[kt] = load_frag(qrow + kt * 32, lane);
    }

    v8f o[4];
#pragma unroll
    for (int nf = 0; nf < 4; ++nf) o[nf] = v8f_zero();
    float mi[8], li[8];
#pragma unroll
    for (int r = 0; r < 8; ++r) { mi[r] = -3.0e38f; li[r] = 0.0f; }

    const int kkey = t >> 2;          // 0..63
    const int kd   = (t & 3) * 16;    // 0,16,32,48

    for (int j = 0; j < Sc; j += 64) {
        // --- stage K via async-to-LDS, V transposed manually ---
        const __bf16* gk = Kb + ((size_t)(b * Sc + j + kkey) * Ec + h * DKc + kd);
        async_copy_b128(&sK[kkey * 64 + kd],     gk);
        async_copy_b128(&sK[kkey * 64 + kd + 8], gk + 8);

        const __bf16* gv = Vb + ((size_t)(b * Sc + j + kkey) * Ec + h * DKc + kd);
        const v8bf v0 = *(const v8bf*)gv;
        const v8bf v1 = *(const v8bf*)(gv + 8);
#pragma unroll
        for (int i = 0; i < 8; ++i) {
            sV[(kd + i) * 64 + kkey]     = v0[i];
            sV[(kd + 8 + i) * 64 + kkey] = v1[i];
        }
        wait_async0();
        __syncthreads();

        // --- scores S = Q @ K^T (16 x 64 per wave) ---
        v8f s[4];
#pragma unroll
        for (int nf = 0; nf < 4; ++nf) s[nf] = v8f_zero();
#pragma unroll
        for (int kt = 0; kt < 2; ++kt)
#pragma unroll
            for (int nf = 0; nf < 4; ++nf) {
                const v16bf bk =
                    load_frag(&sK[(nf * 16 + cl) * 64 + kt * 32], lane);
                s[nf] = wmma_bf16(qa[kt], bk, s[nf]);
            }

        // --- scale + mask ---
#pragma unroll
        for (int nf = 0; nf < 4; ++nf) {
            const int col   = j + nf * 16 + cl;
            const bool dead = (mask[b * Sc + col] == 0);
#pragma unroll
            for (int r = 0; r < 8; ++r) {
                const float sv = s[nf][r] * 0.125f; // 1/sqrt(64)
                s[nf][r] = dead ? -1.0e9f : sv;
            }
        }

        // --- online softmax (row stats live in 16-lane groups) ---
#pragma unroll
        for (int r = 0; r < 8; ++r) {
            float rm = fmaxf(fmaxf(s[0][r], s[1][r]), fmaxf(s[2][r], s[3][r]));
#pragma unroll
            for (int off = 1; off < 16; off <<= 1)
                rm = fmaxf(rm, __shfl_xor(rm, off, 32));
            const float mnew = fmaxf(mi[r], rm);
            const float corr = __expf(mi[r] - mnew);
            mi[r] = mnew;
            li[r] *= corr;
#pragma unroll
            for (int nf = 0; nf < 4; ++nf) o[nf][r] *= corr;
            float rsum = 0.0f;
#pragma unroll
            for (int nf = 0; nf < 4; ++nf) {
                const float p = __expf(s[nf][r] - mnew);
                s[nf][r] = p;
                rsum += p;
            }
#pragma unroll
            for (int off = 1; off < 16; off <<= 1)
                rsum += __shfl_xor(rsum, off, 32);
            li[r] += rsum;
        }

        // --- P: C-layout -> A-layout via wave-private LDS ---
        __bf16* pw = &sP[w * 16 * 64];
#pragma unroll
        for (int nf = 0; nf < 4; ++nf)
#pragma unroll
            for (int r = 0; r < 8; ++r)
                pw[(r + 8 * g) * 64 + nf * 16 + cl] = (__bf16)s[nf][r];
        v16bf pa[2];
#pragma unroll
        for (int kt = 0; kt < 2; ++kt)
            pa[kt] = load_frag(&pw[cl * 64 + kt * 32], lane);

        // --- O += P @ V ---
#pragma unroll
        for (int kt = 0; kt < 2; ++kt)
#pragma unroll
            for (int nf = 0; nf < 4; ++nf) {
                const v16bf bv =
                    load_frag(&sV[(nf * 16 + cl) * 64 + kt * 32], lane);
                o[nf] = wmma_bf16(pa[kt], bv, o[nf]);
            }
        __syncthreads();
    }

    // --- normalize + write ctx (bf16) ---
#pragma unroll
    for (int nf = 0; nf < 4; ++nf)
#pragma unroll
        for (int r = 0; r < 8; ++r) {
            const int row   = q0 + r + 8 * g;
            const float val = o[nf][r] / li[r];
            ctx[(size_t)(b * Sc + row) * Ec + h * DKc + nf * 16 + cl] =
                (__bf16)val;
        }
}

// ---------------------------------------------------------------------------
// Host-side launch
// ---------------------------------------------------------------------------
extern "C" void kernel_launch(void* const* d_in, const int* in_sizes, int n_in,
                              void* d_out, int out_size, void* d_ws, size_t ws_size,
                              hipStream_t stream) {
    const float* x      = (const float*)d_in[0];
    const float* Wq     = (const float*)d_in[1];
    const float* bq     = (const float*)d_in[2];
    const float* Wk     = (const float*)d_in[3];
    const float* bk     = (const float*)d_in[4];
    const float* Wv     = (const float*)d_in[5];
    const float* bv     = (const float*)d_in[6];
    const float* Wo     = (const float*)d_in[7];
    const float* bo     = (const float*)d_in[8];
    const float* W1     = (const float*)d_in[9];
    const float* b1     = (const float*)d_in[10];
    const float* W2     = (const float*)d_in[11];
    const float* b2     = (const float*)d_in[12];
    const float* alpha1 = (const float*)d_in[13];
    const float* beta1  = (const float*)d_in[14];
    const float* alpha2 = (const float*)d_in[15];
    const float* beta2  = (const float*)d_in[16];
    const int*   maskp  = (const int*)d_in[17];
    float*       out    = (float*)d_out;

    // workspace layout
    char* p = (char*)d_ws;
    __bf16* WqT = (__bf16*)p; p += (size_t)Ec * Ec * 2;
    __bf16* WkT = (__bf16*)p; p += (size_t)Ec * Ec * 2;
    __bf16* WvT = (__bf16*)p; p += (size_t)Ec * Ec * 2;
    __bf16* WoT = (__bf16*)p; p += (size_t)Ec * Ec * 2;
    __bf16* W1T = (__bf16*)p; p += (size_t)Ec * DFFc * 2;
    __bf16* W2T = (__bf16*)p; p += (size_t)DFFc * Ec * 2;
    __bf16* x2b = (__bf16*)p; p += (size_t)Mtok * Ec * 2;
    __bf16* Qb  = (__bf16*)p; p += (size_t)Mtok * Ec * 2;
    __bf16* Kb  = (__bf16*)p; p += (size_t)Mtok * Ec * 2;
    __bf16* Vb  = (__bf16*)p; p += (size_t)Mtok * Ec * 2;
    __bf16* Cb  = (__bf16*)p; p += (size_t)Mtok * Ec * 2;
    float*  x1f = (float*)p;  p += (size_t)Mtok * Ec * 4;
    __bf16* hb  = (__bf16*)p; p += (size_t)Mtok * DFFc * 2;

    const dim3 blk(256);

    // 1. weight convert + transpose
    w_transpose_kernel<<<(Ec * Ec) / 256, blk, 0, stream>>>(Wq, WqT, Ec, Ec);
    w_transpose_kernel<<<(Ec * Ec) / 256, blk, 0, stream>>>(Wk, WkT, Ec, Ec);
    w_transpose_kernel<<<(Ec * Ec) / 256, blk, 0, stream>>>(Wv, WvT, Ec, Ec);
    w_transpose_kernel<<<(Ec * Ec) / 256, blk, 0, stream>>>(Wo, WoT, Ec, Ec);
    w_transpose_kernel<<<(Ec * DFFc) / 256, blk, 0, stream>>>(W1, W1T, Ec, DFFc);
    w_transpose_kernel<<<(DFFc * Ec) / 256, blk, 0, stream>>>(W2, W2T, DFFc, Ec);

    // 2. LN1
    layernorm_kernel<<<Mtok, blk, 0, stream>>>(x, alpha1, beta1, x2b);

    // 3. QKV projections
    const dim3 gE(Ec / 128, Mtok / 128);
    gemm_bf16_kernel<0><<<gE, blk, 0, stream>>>(x2b, WqT, bq, nullptr, nullptr, Qb,
                                                Mtok, Ec, Ec);
    gemm_bf16_kernel<0><<<gE, blk, 0, stream>>>(x2b, WkT, bk, nullptr, nullptr, Kb,
                                                Mtok, Ec, Ec);
    gemm_bf16_kernel<0><<<gE, blk, 0, stream>>>(x2b, WvT, bv, nullptr, nullptr, Vb,
                                                Mtok, Ec, Ec);

    // 4. attention
    flash_attn_kernel<<<dim3(Bc * Hc, Sc / 128), blk, 0, stream>>>(Qb, Kb, Vb,
                                                                   maskp, Cb);

    // 5. output projection + residual
    gemm_bf16_kernel<2><<<gE, blk, 0, stream>>>(Cb, WoT, bo, x, x1f, nullptr,
                                                Mtok, Ec, Ec);

    // 6. LN2
    layernorm_kernel<<<Mtok, blk, 0, stream>>>(x1f, alpha2, beta2, x2b);

    // 7. FFN
    gemm_bf16_kernel<1><<<dim3(DFFc / 128, Mtok / 128), blk, 0, stream>>>(
        x2b, W1T, b1, nullptr, nullptr, hb, Mtok, DFFc, Ec);
    gemm_bf16_kernel<2><<<gE, blk, 0, stream>>>(hb, W2T, b2, x1f, out, nullptr,
                                                Mtok, Ec, DFFc);
}
